// HypercubeLinear_58248346469061
// MI455X (gfx1250) — matrix operand order repaired and checked by reading, not compiled, not run on hardware
//
#include <hip/hip_runtime.h>

typedef __attribute__((ext_vector_type(16))) __bf16 v16bf;
typedef __attribute__((ext_vector_type(8)))  float  v8f;

#define KPITCH 68   // dword pitch for one 128-wide bf16 row (64 data dwords + 4 pad)

union FragU { uint4 u[2]; v16bf v; };

// Pack two fp32 -> two bf16 (truncate) in a single v_perm_b32:
// dst[15:0] = a[31:16], dst[31:16] = b[31:16]
__device__ __forceinline__ unsigned pack_bf16(float a, float b) {
  return __builtin_amdgcn_perm(__float_as_uint(b), __float_as_uint(a), 0x07060302u);
}

__global__ __launch_bounds__(256) void hypercube_wmma(
    const float* __restrict__ x,
    const float* __restrict__ hweight,
    const float* __restrict__ hbias,
    float* __restrict__ out)
{
  __shared__ unsigned lds[2 * 128 * KPITCH];
  unsigned* aLds = lds;                    // x tile, bf16-packed [row][kpair]
  unsigned* bLds = lds + 128 * KPITCH;     // weight block [orow(n)][kpair]

  const int mtile = blockIdx.x;   // 0..31 -> 128 rows of x
  const int oblk  = blockIdx.y;   // 0..31 -> 128 output cols
  const int tid   = threadIdx.x;
  const int lane  = tid & 31;
  const int wave  = tid >> 5;
  const int wm    = wave >> 2;    // 0..1 : which 64-row half
  const int wn    = wave & 3;     // 0..3 : which 32-col strip
  const int l16   = lane & 15;
  const int khalf = lane >> 4;

  const int m0 = mtile * 128;

  v8f acc[4][2];
  #pragma unroll
  for (int tm = 0; tm < 4; ++tm)
    #pragma unroll
    for (int tn = 0; tn < 2; ++tn)
      acc[tm][tn] = (v8f){0.f, 0.f, 0.f, 0.f, 0.f, 0.f, 0.f, 0.f};

  float biasv[2];
  #pragma unroll
  for (int tn = 0; tn < 2; ++tn)
    biasv[tn] = hbias[oblk * 128 + wn * 32 + tn * 16 + l16];

  for (int nb = 0; nb < 6; ++nb) {
    // neighbor input block and its weight block: W[oblk,iblk] = hweight[nb][iblk]
    const int iblk = (nb == 0) ? oblk : (oblk ^ (1 << (nb - 1)));
    const float* __restrict__ wsrc = hweight + ((size_t)nb * 32 + iblk) * (128 * 128);
    const float* __restrict__ xsrc = x + (size_t)m0 * 4096 + (size_t)iblk * 128;

    if (nb) __syncthreads();
    // Stage 128x128 fp32 tiles -> bf16 LDS. 256 threads * 16 float4 each.
    #pragma unroll
    for (int i = 0; i < 16; ++i) {
      const int idx = tid + i * 256;   // float4 index within tile
      const int row = idx >> 5;        // 32 float4 per 128-wide row
      const int c4  = idx & 31;
      float4 xa = *(const float4*)(xsrc + (size_t)row * 4096 + c4 * 4);
      aLds[row * KPITCH + c4 * 2    ] = pack_bf16(xa.x, xa.y);
      aLds[row * KPITCH + c4 * 2 + 1] = pack_bf16(xa.z, xa.w);
      float4 wa = *(const float4*)(wsrc + row * 128 + c4 * 4);
      bLds[row * KPITCH + c4 * 2    ] = pack_bf16(wa.x, wa.y);
      bLds[row * KPITCH + c4 * 2 + 1] = pack_bf16(wa.z, wa.w);
    }

    // Prefetch next neighbor's tiles (L2 -> near caches) under this tile's WMMAs.
    // No VGPR staging cost; each lane prefetches 128B lines covering both 64KB tiles.
    if (nb < 5) {
      const int niblk = (nb == 4) ? (oblk ^ 16) : (oblk ^ (1 << nb)); // iblk for nb+1
      const float* xn = x + (size_t)m0 * 4096 + (size_t)niblk * 128;
      const float* wn_ = hweight + ((size_t)(nb + 1) * 32 + niblk) * (128 * 128);
      #pragma unroll
      for (int i = 0; i < 2; ++i) {
        const int j = tid + i * 256;            // 128B line index, 0..511
        __builtin_prefetch(xn + (size_t)(j >> 2) * 4096 + (j & 3) * 32, 0, 3);
        __builtin_prefetch(wn_ + (size_t)j * 32, 0, 3);
      }
    }
    __syncthreads();

    // K = 128 in four 16x16x32 bf16 WMMA steps
    #pragma unroll
    for (int kk = 0; kk < 4; ++kk) {
      const int kp = kk * 16;          // kpair base (K0 = kk*32)
      FragU a[4], b[2];
      #pragma unroll
      for (int tm = 0; tm < 4; ++tm) {
        // A 16x32: lanes 0-15 rows M, K 0-7 & 16-23; lanes 16-31 K 8-15 & 24-31
        const int row = wm * 64 + tm * 16 + l16;
        const unsigned* p = &aLds[row * KPITCH + kp + khalf * 4];
        a[tm].u[0] = *(const uint4*)p;
        a[tm].u[1] = *(const uint4*)(p + 8);
      }
      #pragma unroll
      for (int tn = 0; tn < 2; ++tn) {
        // B 32x16: lane = N col; lanes 0-15 K 0-7 & 16-23; lanes 16-31 K 8-15 & 24-31
        const int n = wn * 32 + tn * 16 + l16;
        const unsigned* p = &bLds[n * KPITCH + kp + khalf * 4];
        b[tn].u[0] = *(const uint4*)p;
        b[tn].u[1] = *(const uint4*)(p + 8);
      }
      #pragma unroll
      for (int tm = 0; tm < 4; ++tm)
        #pragma unroll
        for (int tn = 0; tn < 2; ++tn)
          acc[tm][tn] = __builtin_amdgcn_wmma_f32_16x16x32_bf16(
              false, a[tm].v, false, b[tn].v, (short)0, acc[tm][tn],
              false, false);
    }
  }

  // Epilogue: bias add + fp32 store. C layout: N = lane%16, M = r + 8*(lane/16).
  #pragma unroll
  for (int tm = 0; tm < 4; ++tm) {
    #pragma unroll
    for (int tn = 0; tn < 2; ++tn) {
      const size_t col = (size_t)oblk * 128 + wn * 32 + tn * 16 + l16;
      #pragma unroll
      for (int r = 0; r < 8; ++r) {
        const size_t row = (size_t)m0 + wm * 64 + tm * 16 + khalf * 8 + r;
        out[row * 4096 + col] = acc[tm][tn][r] + biasv[tn];
      }
    }
  }
}

extern "C" void kernel_launch(void* const* d_in, const int* in_sizes, int n_in,
                              void* d_out, int out_size, void* d_ws, size_t ws_size,
                              hipStream_t stream) {
  const float* x       = (const float*)d_in[0];   // (4096, 4096) f32
  const float* hweight = (const float*)d_in[1];   // (6, 32, 128, 128) f32
  const float* hbias   = (const float*)d_in[2];   // (4096,) f32
  float* out           = (float*)d_out;           // (4096, 4096) f32

  dim3 grid(32, 32);   // (M tiles, output blocks)
  hypercube_wmma<<<grid, 256, 0, stream>>>(x, hweight, hbias, out);
}
